// TokenUOTBlock_72533407695528
// MI455X (gfx1250) — matrix operand order repaired and checked by reading, not compiled, not run on hardware
//
#include <hip/hip_runtime.h>
#include <hip/hip_bf16.h>

// ---- problem dims fixed by the reference ----
#define N_TOK 1280
#define DM    512
#define NH    16
#define DH    32
#define DFF   2048
#define KITER 20

typedef __attribute__((ext_vector_type(16))) _Float16 v16h;
typedef __attribute__((ext_vector_type(8)))  _Float16 v8h;
typedef __attribute__((ext_vector_type(8)))  float    v8f;

// ---------------- block reductions (blockDim.x == 256) ----------------
__device__ __forceinline__ float blk_sum(float v, float* red) {
  const int tid = threadIdx.x;
  __syncthreads();
  red[tid] = v;
  __syncthreads();
  for (int s = 128; s > 0; s >>= 1) {
    if (tid < s) red[tid] += red[tid + s];
    __syncthreads();
  }
  return red[0];
}
__device__ __forceinline__ float blk_max(float v, float* red) {
  const int tid = threadIdx.x;
  __syncthreads();
  red[tid] = v;
  __syncthreads();
  for (int s = 128; s > 0; s >>= 1) {
    if (tid < s) red[tid] = fmaxf(red[tid], red[tid + s]);
    __syncthreads();
  }
  return red[0];
}

// ---------------- affine LayerNorm over D=512, one block per row ----------------
__global__ __launch_bounds__(256)
void k_layernorm(const float* __restrict__ x, const float* __restrict__ g,
                 const float* __restrict__ b, float* __restrict__ y) {
  __shared__ float red[256];
  const int row = blockIdx.x;
  const float* xr = x + (size_t)row * DM;
  float* yr = y + (size_t)row * DM;
  float s = 0.f, s2 = 0.f;
  for (int i = threadIdx.x; i < DM; i += 256) { float t = xr[i]; s += t; s2 += t * t; }
  const float mean = blk_sum(s, red) * (1.0f / DM);
  const float ms   = blk_sum(s2, red) * (1.0f / DM);
  const float inv  = rsqrtf(ms - mean * mean + 1e-5f);
  for (int i = threadIdx.x; i < DM; i += 256)
    yr[i] = (xr[i] - mean) * inv * g[i] + b[i];
}

// ---------------- WMMA GEMM: C[M,N] = A[M,K] (f32) @ W[N,K]^T (f32) ----------------
// block = 256 thr (8 waves); block tile 64(M) x 32(N); K step 32; f16 in, f32 acc.
// grid = (N/32, M/64, Z); per-z strides for batched-per-head GEMMs.
__global__ __launch_bounds__(256)
void k_gemm_xwT(const float* __restrict__ A, const float* __restrict__ W,
                float* __restrict__ C, int M, int N, int K,
                long long sA, long long sW, long long sC) {
  __shared__ __attribute__((aligned(16))) _Float16 As[64 * 40];
  __shared__ __attribute__((aligned(16))) _Float16 Bs[32 * 40];
  A += (long long)blockIdx.z * sA;
  W += (long long)blockIdx.z * sW;
  C += (long long)blockIdx.z * sC;
  const int tid  = threadIdx.x;
  const int lane = tid & 31;
  const int wave = tid >> 5;
  const int m0 = blockIdx.y * 64 + (wave >> 1) * 16;
  const int n0 = blockIdx.x * 32 + (wave & 1) * 16;
  const int ar = tid >> 2, ac = (tid & 3) * 8;   // A stage: 64 x 32, 8 f32/thread
  const int br = tid >> 3, bc = (tid & 7) * 4;   // B stage: 32 x 32, 4 f32/thread
  const int arow = blockIdx.y * 64 + ar;
  const int brow = blockIdx.x * 32 + br;
  const int lrow  = lane & 15;
  const int khalf = lane >> 4;
  v8f acc = {};
  for (int k0 = 0; k0 < K; k0 += 32) {
    __syncthreads();
    {
      const float* src = A + (size_t)arow * K + k0 + ac;
      _Float16* dst = As + ar * 40 + ac;
#pragma unroll
      for (int i = 0; i < 8; ++i) dst[i] = (_Float16)src[i];
    }
    {
      const float* src = W + (size_t)brow * K + k0 + bc;
      _Float16* dst = Bs + br * 40 + bc;
#pragma unroll
      for (int i = 0; i < 4; ++i) dst[i] = (_Float16)src[i];
    }
    __syncthreads();
    // A frag (16x32): lanes 0-15 rows M, elems 0..7 -> K kb..kb+7, 8..15 -> K kb+16..kb+23 (kb = khalf*8)
    const _Float16* ap = As + ((wave >> 1) * 16 + lrow) * 40 + khalf * 8;
    const v8h alo = *(const v8h*)(ap);
    const v8h ahi = *(const v8h*)(ap + 16);
    // B frag (32x16): lanes 0-15 cols N, elems 0..7 -> K kb..kb+7, 8..15 -> K kb+16..kb+23 (kb = khalf*8; Bs holds B^T: [n][k])
    const _Float16* bp = Bs + ((wave & 1) * 16 + lrow) * 40 + khalf * 8;
    const v8h blo = *(const v8h*)(bp);
    const v8h bhi = *(const v8h*)(bp + 16);
    v16h af, bf;
#pragma unroll
    for (int i = 0; i < 8; ++i) {
      af[i] = alo[i]; af[i + 8] = ahi[i];
      bf[i] = blo[i]; bf[i + 8] = bhi[i];
    }
    acc = __builtin_amdgcn_wmma_f32_16x16x32_f16(false, af, false, bf,
                                                 (short)0, acc, false, false);
  }
  const int col   = n0 + lrow;
  const int rbase = m0 + khalf * 8;
#pragma unroll
  for (int v = 0; v < 8; ++v)
    C[(size_t)(rbase + v) * N + col] = acc[v];
}

// ------------- per-(h,n) wave: QK plain layernorm over dh=32 + reshape; V -> V^T -------------
__global__ __launch_bounds__(256)
void k_qkln_reshape(const float* __restrict__ Qb, const float* __restrict__ Kb,
                    const float* __restrict__ Vb, float* __restrict__ Qn,
                    float* __restrict__ Kn, float* __restrict__ Vt) {
  const int gw   = blockIdx.x * 8 + (threadIdx.x >> 5);
  const int lane = threadIdx.x & 31;
  const int h = gw / N_TOK;
  const int n = gw - h * N_TOK;
  const size_t src = (size_t)n * DM + h * DH + lane;
  {
    float q = Qb[src];
    float s = q;
    for (int m = 16; m > 0; m >>= 1) s += __shfl_xor(s, m, 32);
    const float d = q - s * (1.0f / 32.0f);
    float v = d * d;
    for (int m = 16; m > 0; m >>= 1) v += __shfl_xor(v, m, 32);
    Qn[((size_t)h * N_TOK + n) * DH + lane] = d * rsqrtf(v * (1.0f / 32.0f) + 1e-5f);
  }
  {
    float k = Kb[src];
    float s = k;
    for (int m = 16; m > 0; m >>= 1) s += __shfl_xor(s, m, 32);
    const float d = k - s * (1.0f / 32.0f);
    float v = d * d;
    for (int m = 16; m > 0; m >>= 1) v += __shfl_xor(v, m, 32);
    Kn[((size_t)h * N_TOK + n) * DH + lane] = d * rsqrtf(v * (1.0f / 32.0f) + 1e-5f);
  }
  Vt[((size_t)h * DH + lane) * N_TOK + n] = Vb[src];
}

// ------------- logK = (scores/sqrt(dh) + w_rel - w_dist*d2/r0^2) / eps -------------
__global__ __launch_bounds__(256)
void k_build_logK(float* __restrict__ logK, const float* __restrict__ wrel,
                  const float* __restrict__ wdist, const float* __restrict__ x) {
  const size_t idx = (size_t)blockIdx.x * 256 + threadIdx.x;
  const int m = (int)(idx % N_TOK);
  const size_t t = idx / N_TOK;
  const int n = (int)(t % N_TOK);
  const int h = (int)(t / N_TOK);
  const float dx = x[n * 3 + 0] - x[m * 3 + 0];
  const float dy = x[n * 3 + 1] - x[m * 3 + 1];
  const float dz = x[n * 3 + 2] - x[m * 3 + 2];
  const float d2 = dx * dx + dy * dy + dz * dz;
  const float eps = 0.5f * (float)(1 << (h >> 2));
  const float sc = 0.17677669529663687f; // 1/sqrt(32)
  logK[idx] = (logK[idx] * sc + wrel[idx] - wdist[h] * d2 * 0.01f) / eps;
}

__global__ __launch_bounds__(256)
void k_zero(float* __restrict__ p, int n) {
  const int i = blockIdx.x * 256 + threadIdx.x;
  if (i < n) p[i] = 0.0f;
}

// ------------- Sinkhorn row pass: lu[h,n] = fi*(log mu - LSE_m(logK + lv)) -------------
__global__ __launch_bounds__(256)
void k_sink_row(const float* __restrict__ logK, const float* __restrict__ mu,
                const float* __restrict__ lv, float* __restrict__ lu) {
  __shared__ float red[256];
  const int h = blockIdx.y, n = blockIdx.x, tid = threadIdx.x;
  const float* row = logK + ((size_t)h * N_TOK + n) * N_TOK;
  const float* lvh = lv + h * N_TOK;
  float v[5];
  float mx = -3.0e38f;
#pragma unroll
  for (int i = 0; i < 5; ++i) {
    const int m = tid + i * 256;
    const float t = row[m] + lvh[m];
    v[i] = t;
    mx = fmaxf(mx, t);
  }
  const float bmx = blk_max(mx, red);
  float s = 0.f;
#pragma unroll
  for (int i = 0; i < 5; ++i) s += __expf(v[i] - bmx);
  const float bs = blk_sum(s, red);
  if (tid == 0) {
    const float eps = 0.5f * (float)(1 << (h >> 2));
    const float fi = 1.0f / (1.0f + eps);
    const float lmu = __logf(fmaxf(mu[h * N_TOK + n], 1e-8f));
    lu[h * N_TOK + n] = fi * (lmu - (bmx + __logf(bs)));
  }
}

// ------------- Sinkhorn col pass: thread-per-column streaming LSE (coalesced) -------------
__global__ __launch_bounds__(256)
void k_sink_col(const float* __restrict__ logK, const float* __restrict__ nu,
                const float* __restrict__ lu, float* __restrict__ lv) {
  __shared__ float lus[N_TOK];
  const int h = blockIdx.y;
  const int m = blockIdx.x * 256 + threadIdx.x;
  for (int i = threadIdx.x; i < N_TOK; i += 256) lus[i] = lu[h * N_TOK + i];
  __syncthreads();
  const float* base = logK + (size_t)h * N_TOK * N_TOK + m;
  float mx = -3.0e38f, s = 0.f;
  for (int n = 0; n < N_TOK; ++n) {
    const float t = base[(size_t)n * N_TOK] + lus[n];
    const float nm = fmaxf(mx, t);
    s = s * __expf(mx - nm) + __expf(t - nm);
    mx = nm;
  }
  const float eps = 0.5f * (float)(1 << (h >> 2));
  const float fi = 1.0f / (1.0f + eps);
  const float lnu = __logf(fmaxf(nu[h * N_TOK + m], 1e-8f));
  lv[h * N_TOK + m] = fi * (lnu - (mx + __logf(s)));
}

// ------------- P = exp(lu + logK + lv) in place; rowsum(P); P @ x_res -------------
__global__ __launch_bounds__(256)
void k_compute_P(float* __restrict__ logK, const float* __restrict__ lu,
                 const float* __restrict__ lv, const float* __restrict__ x,
                 float* __restrict__ rowsum, float* __restrict__ xcnum) {
  __shared__ float red[256];
  const int h = blockIdx.y, n = blockIdx.x, tid = threadIdx.x;
  float* row = logK + ((size_t)h * N_TOK + n) * N_TOK;
  const float* lvh = lv + h * N_TOK;
  const float lun = lu[h * N_TOK + n];
  float s = 0.f, sx = 0.f, sy = 0.f, sz = 0.f;
#pragma unroll
  for (int i = 0; i < 5; ++i) {
    const int m = tid + i * 256;
    const float p = __expf(lun + row[m] + lvh[m]);
    row[m] = p;
    s += p;
    sx += p * x[m * 3 + 0];
    sy += p * x[m * 3 + 1];
    sz += p * x[m * 3 + 2];
  }
  const float bs  = blk_sum(s, red);
  const float bsx = blk_sum(sx, red);
  const float bsy = blk_sum(sy, red);
  const float bsz = blk_sum(sz, red);
  if (tid == 0) {
    rowsum[h * N_TOK + n] = bs;
    xcnum[((size_t)h * N_TOK + n) * 3 + 0] = bsx;
    xcnum[((size_t)h * N_TOK + n) * 3 + 1] = bsy;
    xcnum[((size_t)h * N_TOK + n) * 3 + 2] = bsz;
  }
}

// ------------- o = sigmoid(G) * attn, gathered back to (N, D) -------------
__global__ __launch_bounds__(256)
void k_gate(const float* __restrict__ G, const float* __restrict__ attn,
            float* __restrict__ o) {
  const size_t idx = (size_t)blockIdx.x * 256 + threadIdx.x;
  const int n = (int)(idx / DM);
  const int c = (int)(idx % DM);
  const int h = c >> 5, d = c & 31;
  const float g = G[idx];
  const float sg = 1.0f / (1.0f + __expf(-g));
  o[idx] = sg * attn[((size_t)h * N_TOK + n) * DH + d];
}

__global__ __launch_bounds__(256)
void k_add2(const float* __restrict__ a, const float* __restrict__ b,
            float* __restrict__ y, int n) {
  const int i = blockIdx.x * 256 + threadIdx.x;
  if (i < n) y[i] = a[i] + b[i];
}
__global__ __launch_bounds__(256)
void k_addip(float* __restrict__ y, const float* __restrict__ a, int n) {
  const int i = blockIdx.x * 256 + threadIdx.x;
  if (i < n) y[i] += a[i];
}
__global__ __launch_bounds__(256)
void k_silu_mul(float* __restrict__ a, const float* __restrict__ b, int n) {
  const int i = blockIdx.x * 256 + threadIdx.x;
  if (i < n) {
    const float x = a[i];
    a[i] = (x / (1.0f + __expf(-x))) * b[i];
  }
}

// ------------- coordinate update: x + sum_h gamma[h]*(x - xc_h) -------------
__global__ __launch_bounds__(256)
void k_xout(const float* __restrict__ x, const float* __restrict__ gamma,
            const float* __restrict__ rowsum, const float* __restrict__ xcnum,
            float* __restrict__ xout) {
  const int n = blockIdx.x * 256 + threadIdx.x;
  if (n >= N_TOK) return;
#pragma unroll
  for (int c = 0; c < 3; ++c) {
    const float xr = x[n * 3 + c];
    float acc = xr;
    for (int h = 0; h < NH; ++h) {
      const float cs = xcnum[((size_t)h * N_TOK + n) * 3 + c] /
                       (rowsum[h * N_TOK + n] + 1e-8f);
      acc += gamma[h] * (xr - cs);
    }
    xout[n * 3 + c] = acc;
  }
}

// ======================================================================
extern "C" void kernel_launch(void* const* d_in, const int* in_sizes, int n_in,
                              void* d_out, int out_size, void* d_ws, size_t ws_size,
                              hipStream_t stream) {
  (void)in_sizes; (void)n_in; (void)out_size; (void)ws_size;
  const float* h        = (const float*)d_in[0];
  const float* x_res    = (const float*)d_in[1];
  const float* mu       = (const float*)d_in[2];
  const float* nu       = (const float*)d_in[3];
  const float* w_rel    = (const float*)d_in[4];
  const float* w_dist   = (const float*)d_in[5];
  const float* ln_ag    = (const float*)d_in[7];
  const float* ln_ab    = (const float*)d_in[8];
  const float* Wq       = (const float*)d_in[9];
  const float* Wk       = (const float*)d_in[10];
  const float* Wv       = (const float*)d_in[11];
  const float* Wg       = (const float*)d_in[12];
  const float* Wo       = (const float*)d_in[13];
  const float* gamma    = (const float*)d_in[14];
  const float* ln_fg    = (const float*)d_in[15];
  const float* ln_fb    = (const float*)d_in[16];
  const float* W1       = (const float*)d_in[17];
  const float* W2       = (const float*)d_in[18];
  const float* W3       = (const float*)d_in[19];

  // output slices: [h: N*D][x_out: N*3][lu: H*N][lv: H*N]
  float* out_h  = (float*)d_out;
  float* out_x  = out_h + (size_t)N_TOK * DM;
  float* out_lu = out_x + (size_t)N_TOK * 3;
  float* out_lv = out_lu + (size_t)NH * N_TOK;

  // workspace slices (floats)
  float* ws = (float*)d_ws;
  size_t cur = 0;
  const size_t ND = (size_t)N_TOK * DM;     // 655360
  float* h_n  = ws + cur; cur += ND;
  float* Qb   = ws + cur; cur += ND;
  float* Kb   = ws + cur; cur += ND;
  float* Vb   = ws + cur; cur += ND;
  float* Gb   = ws + cur; cur += ND;
  float* Qn   = ws + cur; cur += ND;        // (H,N,dh)
  float* Kn   = ws + cur; cur += ND;        // (H,N,dh)
  float* Vt   = ws + cur; cur += ND;        // (H,dh,N)
  float* attn = ws + cur; cur += ND;        // (H,N,dh)
  float* o    = ws + cur; cur += ND;
  float* tmp  = ws + cur; cur += ND;
  float* hf   = ws + cur; cur += ND;
  float* rsum = ws + cur; cur += (size_t)NH * N_TOK;
  float* xcn  = ws + cur; cur += (size_t)NH * N_TOK * 3;
  float* ff1  = ws + cur; cur += (size_t)N_TOK * DFF;
  float* ff2  = ws + cur; cur += (size_t)N_TOK * DFF;
  float* logK = ws + cur; cur += (size_t)NH * N_TOK * N_TOK; // 104.8 MB

  const dim3 blk(256);

  // 1) attn pre-LN
  k_layernorm<<<N_TOK, blk, 0, stream>>>(h, ln_ag, ln_ab, h_n);

  // 2) Q,K,V,G projections (WMMA)
  k_gemm_xwT<<<dim3(DM / 32, N_TOK / 64, 1), blk, 0, stream>>>(h_n, Wq, Qb, N_TOK, DM, DM, 0, 0, 0);
  k_gemm_xwT<<<dim3(DM / 32, N_TOK / 64, 1), blk, 0, stream>>>(h_n, Wk, Kb, N_TOK, DM, DM, 0, 0, 0);
  k_gemm_xwT<<<dim3(DM / 32, N_TOK / 64, 1), blk, 0, stream>>>(h_n, Wv, Vb, N_TOK, DM, DM, 0, 0, 0);
  k_gemm_xwT<<<dim3(DM / 32, N_TOK / 64, 1), blk, 0, stream>>>(h_n, Wg, Gb, N_TOK, DM, DM, 0, 0, 0);

  // 3) QK-layernorm + head reshape + V transpose
  k_qkln_reshape<<<(NH * N_TOK) / 8, blk, 0, stream>>>(Qb, Kb, Vb, Qn, Kn, Vt);

  // 4) scores per head (WMMA, batched over grid.z), fused into logK
  k_gemm_xwT<<<dim3(N_TOK / 32, N_TOK / 64, NH), blk, 0, stream>>>(
      Qn, Kn, logK, N_TOK, N_TOK, DH,
      (long long)N_TOK * DH, (long long)N_TOK * DH, (long long)N_TOK * N_TOK);
  k_build_logK<<<(NH * (size_t)N_TOK * N_TOK) / 256, blk, 0, stream>>>(logK, w_rel, w_dist, x_res);

  // 5) Sinkhorn (lu/lv live directly in d_out slices)
  k_zero<<<(2 * NH * N_TOK + 255) / 256, blk, 0, stream>>>(out_lu, 2 * NH * N_TOK);
  for (int it = 0; it < KITER; ++it) {
    k_sink_row<<<dim3(N_TOK, NH), blk, 0, stream>>>(logK, mu, out_lv, out_lu);
    k_sink_col<<<dim3(N_TOK / 256, NH), blk, 0, stream>>>(logK, nu, out_lu, out_lv);
  }

  // 6) P (in place over logK) + rowsum + P @ x_res
  k_compute_P<<<dim3(N_TOK, NH), blk, 0, stream>>>(logK, out_lu, out_lv, x_res, rsum, xcn);

  // 7) attn = P @ V per head (WMMA; Vt rows are output features)
  k_gemm_xwT<<<dim3(DH / 32, N_TOK / 64, NH), blk, 0, stream>>>(
      logK, Vt, attn, N_TOK, DH, N_TOK,
      (long long)N_TOK * N_TOK, (long long)DH * N_TOK, (long long)N_TOK * DH);

  // 8) gate, output projection, residual (h_mid lives in out_h)
  k_gate<<<ND / 256, blk, 0, stream>>>(Gb, attn, o);
  k_gemm_xwT<<<dim3(DM / 32, N_TOK / 64, 1), blk, 0, stream>>>(o, Wo, tmp, N_TOK, DM, DM, 0, 0, 0);
  k_add2<<<(int)(ND / 256), blk, 0, stream>>>(h, tmp, out_h, (int)ND);

  // 9) coordinate update
  k_xout<<<(N_TOK + 255) / 256, blk, 0, stream>>>(x_res, gamma, rsum, xcn, out_x);

  // 10) feed-forward (WMMA x3) with SiLU gating, residual in place
  k_layernorm<<<N_TOK, blk, 0, stream>>>(out_h, ln_fg, ln_fb, hf);
  k_gemm_xwT<<<dim3(DFF / 32, N_TOK / 64, 1), blk, 0, stream>>>(hf, W1, ff1, N_TOK, DFF, DM, 0, 0, 0);
  k_gemm_xwT<<<dim3(DFF / 32, N_TOK / 64, 1), blk, 0, stream>>>(hf, W2, ff2, N_TOK, DFF, DM, 0, 0, 0);
  k_silu_mul<<<(int)(((size_t)N_TOK * DFF) / 256), blk, 0, stream>>>(ff1, ff2, N_TOK * DFF);
  k_gemm_xwT<<<dim3(DM / 32, N_TOK / 64, 1), blk, 0, stream>>>(ff1, W3, tmp, N_TOK, DM, DFF, 0, 0, 0);
  k_addip<<<(int)(ND / 256), blk, 0, stream>>>(out_h, tmp, (int)ND);
}
